// AvgRnnNew_17858474017391
// MI455X (gfx1250) — matrix operand order repaired and checked by reading, not compiled, not run on hardware
//
#include <hip/hip_runtime.h>
#include <hip/hip_bf16.h>
#include <math.h>

// ---------------- problem constants (match reference) ----------------
#define Bc 32
#define Pc 24
#define Sc 48
#define Hc 512
#define Fc 64
#define Dc 576            // H + F (LSTM input width)
#define Oc 16
#define Kx 18             // x K-chunks of 32 (576/32)
#define Kh 16             // h K-chunks of 32 (512/32)
#define NT 128            // N tiles (2048 gates / 16)
#define EPSC 1e-5f

typedef __attribute__((ext_vector_type(16))) __bf16 v16bf;
typedef __attribute__((ext_vector_type(8)))  float  v8f;

union Frag {
  v16bf v;
  unsigned short u[16];
  int4 q[2];
};

static __device__ __forceinline__ unsigned short f2bf(float f) {
  unsigned int x = __float_as_uint(f);
  x += 0x7FFFu + ((x >> 16) & 1u);           // round-to-nearest-even
  return (unsigned short)(x >> 16);
}
static __device__ __forceinline__ float sigm(float x) {
  return 1.0f / (1.0f + expf(-x));
}
static __device__ __forceinline__ v8f zero8() {
  v8f z;
#pragma unroll
  for (int i = 0; i < 8; ++i) z[i] = 0.0f;
  return z;
}

// ============================================================================
// K0: pack W (f32 row-major, width cols) -> bf16 WMMA B-fragments.
// Fragment (ntile, kc): 1 KB; lane L holds 16 contiguous bf16:
//   n = L%16, K = 16*(L/16) + e  ->  W[ntile*16+n][kc*32 + K]
// ============================================================================
__global__ void pack_w_kernel(const float* __restrict__ W,
                              unsigned short* __restrict__ out,
                              int kcTot, int width) {
  long tid = (long)blockIdx.x * blockDim.x + threadIdx.x;
  long total = (long)NT * kcTot * 512;
  if (tid >= total) return;
  int frag  = (int)(tid >> 9);
  int r     = (int)(tid & 511);
  int lane  = r >> 4, e = r & 15;
  int ntile = frag / kcTot, kc = frag % kcTot;
  int n = lane & 15;
  int k = 16 * (lane >> 4) + e;
  int j = ntile * 16 + n;
  int d = kc * 32 + k;
  out[tid] = f2bf(W[(long)j * width + d]);
}

// ============================================================================
// K1: embedding gather + masked attention softmax + weighted pooling.
// One block per (b,p); emits rep directly in bf16 (the GEMMs consume bf16
// anyway), so the sequential scan never converts activations.
// ============================================================================
__global__ void __launch_bounds__(256) attn_kernel(
    const int* __restrict__ x, const unsigned char* __restrict__ xmask,
    const float* __restrict__ xfeat, const float* __restrict__ emb,
    const float* __restrict__ wattn, const float* __restrict__ battn,
    unsigned short* __restrict__ rep16) {
  __shared__ int           sTok[Sc];
  __shared__ float         sScore[Sc];
  __shared__ float         sAlpha[Sc];
  __shared__ unsigned char sMask[Sc];
  int bp   = blockIdx.x;            // b*P + p
  int tid  = threadIdx.x;
  int lane = tid & 31, wave = tid >> 5;

  if (tid < Sc) {
    sTok[tid]  = x[(long)bp * Sc + tid];
    sMask[tid] = xmask[(long)bp * Sc + tid];
  }
  __syncthreads();

  for (int s = wave; s < Sc; s += 8) {
    const float* erow = emb + (long)sTok[s] * Hc;
    float part = 0.f;
    for (int h = lane; h < Hc; h += 32) part += erow[h] * wattn[h];
    for (int off = 16; off; off >>= 1) part += __shfl_xor(part, off, 32);
    if (lane == 0) sScore[s] = part + battn[0];
  }
  __syncthreads();

  float mx = -3.0e38f;
  for (int s = 0; s < Sc; ++s) {
    float sc = sMask[s] ? -1.0e30f : sScore[s];
    mx = fmaxf(mx, sc);
  }
  float sum = 0.f;
  for (int s = 0; s < Sc; ++s) {
    float sc = sMask[s] ? -1.0e30f : sScore[s];
    sum += expf(sc - mx);
  }
  if (tid < Sc) {
    float sc = sMask[tid] ? -1.0e30f : sScore[tid];
    sAlpha[tid] = sMask[tid] ? 0.f : expf(sc - mx) / sum;
  }
  __syncthreads();

  unsigned short* orow = rep16 + (long)bp * Dc;
  const float* frow = xfeat + (long)bp * Sc * Fc;
  for (int d = tid; d < Dc; d += 256) {
    float acc = 0.f;
    if (d < Hc) {
      for (int s = 0; s < Sc; ++s) acc += sAlpha[s] * emb[(long)sTok[s] * Hc + d];
    } else {
      int f = d - Hc;
      for (int s = 0; s < Sc; ++s)
        if (!sMask[s]) acc += frow[s * Fc + f];
    }
    orow[d] = f2bf(acc);
  }
}

// ============================================================================
// K2: all recurrence-free GEMMs in parallel across WGPs.
//  blocks 0..23 : Gx[t] = x_t @ Wih_f.T + (bih_f+bhh_f), stored as f32
//                 C-fragments (lane-major: lane*8+r) for the scan to reload.
//  block  24    : complete backward direction. Only h_bwd_rev[0] reaches the
//                 output (rev-index algebra), and its state is zero, so it is
//                 a single x-GEMM + elementwise with c = sig(i)*tanh(g).
// ============================================================================
__global__ void __launch_bounds__(1024) gemm_x_kernel(
    const unsigned short* __restrict__ rep16,
    const unsigned short* __restrict__ Wihf,   // packed, 18 kc
    const unsigned short* __restrict__ Wib,    // packed, 18 kc
    const float* __restrict__ bih_f, const float* __restrict__ bhh_f,
    const float* __restrict__ bih_b, const float* __restrict__ bhh_b,
    const int* __restrict__ lens,
    float* __restrict__ Gx, float* __restrict__ hB) {
  int tid   = threadIdx.x;
  int lane  = tid & 31;
  int wave  = tid >> 5;               // N-column tile, 0..31
  int nlo   = lane & 15;
  int kbase = (lane >> 4) * 16;

  v8f acc[2][4];
#pragma unroll
  for (int jj = 0; jj < 2; ++jj)
#pragma unroll
    for (int q = 0; q < 4; ++q) acc[jj][q] = zero8();

  bool fwd = (blockIdx.x < Pc);
  int t0, t1;
  if (fwd) {
    t0 = t1 = blockIdx.x;
  } else {
    t0 = lens[nlo] - 1;               // backward reads seq[lens-1]
    t1 = lens[16 + nlo] - 1;
  }
  const unsigned short* r0 = rep16 + ((long)(nlo)      * Pc + t0) * Dc;
  const unsigned short* r1 = rep16 + ((long)(16 + nlo) * Pc + t1) * Dc;
  const unsigned short* W  = fwd ? Wihf : Wib;

  for (int kc = 0; kc < Kx; ++kc) {
    Frag a0, a1;
    a0.q[0] = *(const int4*)(r0 + kc * 32 + kbase);
    a0.q[1] = *(const int4*)(r0 + kc * 32 + 8 + kbase);
    a1.q[0] = *(const int4*)(r1 + kc * 32 + kbase);
    a1.q[1] = *(const int4*)(r1 + kc * 32 + 8 + kbase);
#pragma unroll
    for (int q = 0; q < 4; ++q) {
      Frag b;
      const int4* bp = (const int4*)(W + ((long)((q * 32 + wave) * Kx + kc) << 9)) + lane * 2;
      b.q[0] = bp[0];
      b.q[1] = bp[1];
      acc[0][q] = __builtin_amdgcn_wmma_f32_16x16x32_bf16(
          false, a0.v, false, b.v, (short)0, acc[0][q], false, false);
      acc[1][q] = __builtin_amdgcn_wmma_f32_16x16x32_bf16(
          false, a1.v, false, b.v, (short)0, acc[1][q], false, false);
    }
  }

  if (fwd) {
    int t = blockIdx.x;
#pragma unroll
    for (int q = 0; q < 4; ++q) {
      int j = q * Hc + wave * 16 + nlo;
      float bj = bih_f[j] + bhh_f[j];
#pragma unroll
      for (int jj = 0; jj < 2; ++jj) {
        float* g = Gx + ((((long)t * NT + (q * 32 + wave)) * 2 + jj) << 8) + lane * 8;
        float4 lo, hi;
        lo.x = acc[jj][q][0] + bj; lo.y = acc[jj][q][1] + bj;
        lo.z = acc[jj][q][2] + bj; lo.w = acc[jj][q][3] + bj;
        hi.x = acc[jj][q][4] + bj; hi.y = acc[jj][q][5] + bj;
        hi.z = acc[jj][q][6] + bj; hi.w = acc[jj][q][7] + bj;
        *(float4*)(g)     = lo;
        *(float4*)(g + 4) = hi;
      }
    }
  } else {
    float bias[4];
#pragma unroll
    for (int q = 0; q < 4; ++q) {
      int j = q * Hc + wave * 16 + nlo;
      bias[q] = bih_b[j] + bhh_b[j];
    }
#pragma unroll
    for (int jj = 0; jj < 2; ++jj) {
#pragma unroll
      for (int r = 0; r < 8; ++r) {
        float iv = sigm(acc[jj][0][r] + bias[0]);
        float gv = tanhf(acc[jj][2][r] + bias[2]);
        float ov = sigm(acc[jj][3][r] + bias[3]);
        float c  = iv * gv;             // c0 = 0 -> f-gate term vanishes
        float h  = ov * tanhf(c);
        int mrow = jj * 16 + r + 8 * (lane >> 4);
        int col  = wave * 16 + nlo;
        hB[mrow * Hc + col] = h;
      }
    }
  }
}

// ============================================================================
// K3: the only sequential part. 24 steps; per step: reload precomputed
// x-gates (Gx), add the h@Whh.T recurrence via bf16 WMMA (16 K-chunks),
// elementwise LSTM in f32 registers; h round-trips through a 32KB LDS
// bf16 buffer. c-state never leaves registers.
// ============================================================================
__global__ void __launch_bounds__(1024) lstm_scan_kernel(
    const float* __restrict__ Gx,
    const unsigned short* __restrict__ Whhf,   // packed, 16 kc
    float* __restrict__ hF) {
  __shared__ __align__(16) unsigned short hS[Bc * Hc];  // 32 KB bf16 h state
  int tid   = threadIdx.x;
  int lane  = tid & 31;
  int wave  = tid >> 5;
  int nlo   = lane & 15;
  int kbase = (lane >> 4) * 16;

  for (int i = tid; i < Bc * Hc; i += 1024) hS[i] = 0;

  v8f acc[2][4];
  v8f cst[2];
  cst[0] = zero8();
  cst[1] = zero8();
  __syncthreads();

  const unsigned short* hrow0 = hS + nlo * Hc;
  const unsigned short* hrow1 = hS + (16 + nlo) * Hc;

  for (int t = 0; t < Pc; ++t) {
    // ---- init accumulators from precomputed x-gates (+bias) ----
#pragma unroll
    for (int q = 0; q < 4; ++q)
#pragma unroll
      for (int jj = 0; jj < 2; ++jj) {
        const float* g = Gx + ((((long)t * NT + (q * 32 + wave)) * 2 + jj) << 8) + lane * 8;
        float4 lo = *(const float4*)(g);
        float4 hi = *(const float4*)(g + 4);
        acc[jj][q][0] = lo.x; acc[jj][q][1] = lo.y;
        acc[jj][q][2] = lo.z; acc[jj][q][3] = lo.w;
        acc[jj][q][4] = hi.x; acc[jj][q][5] = hi.y;
        acc[jj][q][6] = hi.z; acc[jj][q][7] = hi.w;
      }

    // ---- recurrent GEMM: h_{t-1} @ Whh.T, K-chunks 0..15 ----
    for (int kc = 0; kc < Kh; ++kc) {
      Frag a0, a1;
      a0.q[0] = *(const int4*)(hrow0 + kc * 32 + kbase);
      a0.q[1] = *(const int4*)(hrow0 + kc * 32 + 8 + kbase);
      a1.q[0] = *(const int4*)(hrow1 + kc * 32 + kbase);
      a1.q[1] = *(const int4*)(hrow1 + kc * 32 + 8 + kbase);
#pragma unroll
      for (int q = 0; q < 4; ++q) {
        Frag b;
        const int4* bp = (const int4*)(Whhf + ((long)((q * 32 + wave) * Kh + kc) << 9)) + lane * 2;
        b.q[0] = bp[0];
        b.q[1] = bp[1];
        acc[0][q] = __builtin_amdgcn_wmma_f32_16x16x32_bf16(
            false, a0.v, false, b.v, (short)0, acc[0][q], false, false);
        acc[1][q] = __builtin_amdgcn_wmma_f32_16x16x32_bf16(
            false, a1.v, false, b.v, (short)0, acc[1][q], false, false);
      }
    }

    __syncthreads();  // all GEMM reads of hS complete

    // ---- element-wise LSTM update (f32, in registers; bias already in Gx) ----
#pragma unroll
    for (int jj = 0; jj < 2; ++jj) {
#pragma unroll
      for (int r = 0; r < 8; ++r) {
        float iv = sigm(acc[jj][0][r]);
        float fv = sigm(acc[jj][1][r]);
        float gv = tanhf(acc[jj][2][r]);
        float ov = sigm(acc[jj][3][r]);
        float c  = fv * cst[jj][r] + iv * gv;
        cst[jj][r] = c;
        float h = ov * tanhf(c);
        int mrow = jj * 16 + r + 8 * (lane >> 4);
        int col  = wave * 16 + nlo;
        hS[mrow * Hc + col] = f2bf(h);
        if (t == Pc - 1) hF[mrow * Hc + col] = h;
      }
    }
    __syncthreads();  // h visible before next step's GEMM
  }
}

// ============================================================================
// K4: last-row mask (lens==P), transpose-reshape z, FC, batch-norm over batch,
// relu, log-softmax over batch. One block, 512 threads = one per (i,o).
// ============================================================================
__global__ void __launch_bounds__(512) head_kernel(
    const float* __restrict__ hF, const float* __restrict__ hB,
    const int* __restrict__ lens, const float* __restrict__ Wfc,
    const float* __restrict__ bfc, const float* __restrict__ gamma,
    const float* __restrict__ beta, float* __restrict__ out) {
  __shared__ float yS[Bc * Oc];
  __shared__ float mS[Oc], sS[Oc], lmax[Oc], lsum[Oc];
  __shared__ int lenS[Bc];
  int tid = threadIdx.x;
  if (tid < Bc) lenS[tid] = lens[tid];
  __syncthreads();

  int i = tid >> 4;       // batch row of z, 0..31
  int o = tid & 15;       // class, 0..15
  // z[i][j] = last[j%32][i*32 + j/32];  last[b] = lens[b]==P ? [hF[b],hB[b]] : 0
  float acc = bfc[o];
  const float* wrow = Wfc + (long)o * (2 * Hc);
  for (int j = 0; j < 2 * Hc; ++j) {
    int bb = j & 31;
    int r  = i * 32 + (j >> 5);
    float v = 0.f;
    if (lenS[bb] == Pc)
      v = (r < Hc) ? hF[bb * Hc + r] : hB[bb * Hc + (r - Hc)];
    acc += v * wrow[j];
  }
  yS[i * Oc + o] = acc;
  __syncthreads();

  if (tid < Oc) {
    float m = 0.f;
    for (int b2 = 0; b2 < Bc; ++b2) m += yS[b2 * Oc + tid];
    m /= (float)Bc;
    float v = 0.f;
    for (int b2 = 0; b2 < Bc; ++b2) {
      float d = yS[b2 * Oc + tid] - m;
      v += d * d;
    }
    v /= (float)Bc;
    mS[tid] = m;
    sS[tid] = rsqrtf(v + EPSC);
  }
  __syncthreads();

  float yn = gamma[o] * (yS[i * Oc + o] - mS[o]) * sS[o] + beta[o];
  yn = fmaxf(yn, 0.f);
  yS[i * Oc + o] = yn;
  __syncthreads();

  if (tid < Oc) {
    float m = -3.0e38f;
    for (int b2 = 0; b2 < Bc; ++b2) m = fmaxf(m, yS[b2 * Oc + tid]);
    float s = 0.f;
    for (int b2 = 0; b2 < Bc; ++b2) s += expf(yS[b2 * Oc + tid] - m);
    lmax[tid] = m;
    lsum[tid] = logf(s);
  }
  __syncthreads();

  out[i * Oc + o] = yS[i * Oc + o] - lmax[o] - lsum[o];
}

// ============================================================================
extern "C" void kernel_launch(void* const* d_in, const int* in_sizes, int n_in,
                              void* d_out, int out_size, void* d_ws, size_t ws_size,
                              hipStream_t stream) {
  (void)in_sizes; (void)n_in; (void)out_size; (void)ws_size;
  const int*           x     = (const int*)d_in[0];
  const unsigned char* xmask = (const unsigned char*)d_in[1];   // jnp bool = u8
  const float*         xfeat = (const float*)d_in[2];
  const int*           lens  = (const int*)d_in[3];
  // d_in[4]=clause, d_in[5]=cls: unused by the reference math
  const float* emb   = (const float*)d_in[6];
  const float* wattn = (const float*)d_in[7];
  const float* battn = (const float*)d_in[8];
  const float* Wih_f = (const float*)d_in[9];
  const float* Whh_f = (const float*)d_in[10];
  const float* bih_f = (const float*)d_in[11];
  const float* bhh_f = (const float*)d_in[12];
  const float* Wih_b = (const float*)d_in[13];
  // Whh_b (d_in[14]) multiplies h0=0 in the only backward step that matters
  const float* bih_b = (const float*)d_in[15];
  const float* bhh_b = (const float*)d_in[16];
  const float* Wfc   = (const float*)d_in[17];
  const float* bfc   = (const float*)d_in[18];
  const float* gamma = (const float*)d_in[19];
  const float* beta  = (const float*)d_in[20];

  // ---- workspace layout (all offsets 1KB-aligned) ----
  char* ws = (char*)d_ws;
  unsigned short* Wihf_pack = (unsigned short*)(ws);              // 2,359,296 B
  unsigned short* Whhf_pack = (unsigned short*)(ws +  2359296);   // 2,097,152 B
  unsigned short* Wib_pack  = (unsigned short*)(ws +  4456448);   // 2,359,296 B
  unsigned short* rep16     = (unsigned short*)(ws +  6815744);   //   884,736 B
  float*          Gx        = (float*)         (ws +  7700480);   // 6,291,456 B
  float*          hF        = (float*)         (ws + 13991936);   //    65,536 B
  float*          hB        = hF + Bc * Hc;                       //    65,536 B
  // total ~ 13.5 MB

  long n1 = (long)NT * Kx * 512;   // Wih_f fragments
  long n2 = (long)NT * Kh * 512;   // Whh_f fragments
  long n3 = (long)NT * Kx * 512;   // Wih_b fragments
  pack_w_kernel<<<(unsigned)((n1 + 255) / 256), 256, 0, stream>>>(Wih_f, Wihf_pack, Kx, Dc);
  pack_w_kernel<<<(unsigned)((n2 + 255) / 256), 256, 0, stream>>>(Whh_f, Whhf_pack, Kh, Hc);
  pack_w_kernel<<<(unsigned)((n3 + 255) / 256), 256, 0, stream>>>(Wih_b, Wib_pack, Kx, Dc);
  attn_kernel<<<Bc * Pc, 256, 0, stream>>>(x, xmask, xfeat, emb, wattn, battn, rep16);
  gemm_x_kernel<<<Pc + 1, 1024, 0, stream>>>(rep16, Wihf_pack, Wib_pack,
                                             bih_f, bhh_f, bih_b, bhh_b, lens, Gx, hB);
  lstm_scan_kernel<<<1, 1024, 0, stream>>>(Gx, Whhf_pack, hF);
  head_kernel<<<1, 512, 0, stream>>>(hF, hB, lens, Wfc, bfc, gamma, beta, (float*)d_out);
}
